// DeepEMD_63428077027631
// MI455X (gfx1250) — matrix operand order repaired and checked by reading, not compiled, not run on hardware
//
#include <hip/hip_runtime.h>
#include <hip/hip_bf16.h>

// Problem constants (match reference)
#define BB 256
#define CC 512
#define NN 144            // H*W = 12*12
#define NT 9              // NN / 16 tiles per dim
#define SINK_EPS 0.05f
#define SINK_ITERS 20
#define TEMPERATURE 12.5f

typedef __attribute__((ext_vector_type(16))) __bf16 v16bf;
typedef __attribute__((ext_vector_type(8)))  float  v8f;

__device__ __forceinline__ unsigned short f2bf(float x) {
    unsigned int u = __float_as_uint(x);
    unsigned int r = (u + 0x7FFFu + ((u >> 16) & 1u)) >> 16;  // round-to-nearest-even
    return (unsigned short)r;
}

// ---------------------------------------------------------------------------
// K1: per-(b,c) mean over HW for proto and query  -> pmean/qmean [B*C]
// ---------------------------------------------------------------------------
__global__ void k_means(const float* __restrict__ proto,
                        const float* __restrict__ query,
                        float* __restrict__ pmean,
                        float* __restrict__ qmean) {
    int idx = blockIdx.x * blockDim.x + threadIdx.x;   // (b*C + c)
    if (idx >= BB * CC) return;
    const float* p = proto + (size_t)idx * NN;
    const float* q = query + (size_t)idx * NN;
    float sp = 0.f, sq = 0.f;
    #pragma unroll 4
    for (int i = 0; i < NN; ++i) { sp += p[i]; sq += q[i]; }
    pmean[idx] = sp * (1.f / NN);
    qmean[idx] = sq * (1.f / NN);
}

// ---------------------------------------------------------------------------
// K2: per-batch prep.  One workgroup (256 threads) per batch:
//  - per-node channel mean, centered L2 norm (via sum / sumsq, one pass)
//  - weight vectors w1,w2 -> normalized -> log marginals
//  - write centered+normalized features, node-major [B,N,C], as bf16
// ---------------------------------------------------------------------------
__global__ void k_prep(const float* __restrict__ proto,
                       const float* __restrict__ query,
                       const float* __restrict__ pmean,
                       const float* __restrict__ qmean,
                       unsigned short* __restrict__ qnT,
                       unsigned short* __restrict__ pnT,
                       float* __restrict__ logr,
                       float* __restrict__ logc) {
    const int b = blockIdx.x;
    const int tid = threadIdx.x;
    __shared__ float muq[NN], ivq[NN], mup[NN], ivp[NN], wr[NN], wc[NN];
    __shared__ float sums[2];

    const size_t baseBC = (size_t)b * CC;
    if (tid < NN) {
        const int n = tid;
        float sq = 0.f, ssq = 0.f, w1 = 0.f;
        float sp = 0.f, ssp = 0.f, w2 = 0.f;
        for (int c = 0; c < CC; ++c) {
            float qv = query[(baseBC + c) * NN + n];
            float pv = proto[(baseBC + c) * NN + n];
            float pm = pmean[baseBC + c];
            float qm = qmean[baseBC + c];
            sq += qv; ssq += qv * qv; w1 += qv * pm;   // weight_vector(query, proto)
            sp += pv; ssp += pv * pv; w2 += pv * qm;   // weight_vector(proto, query)
        }
        float mq = sq * (1.f / CC), mp = sp * (1.f / CC);
        float nq = sqrtf(fmaxf(ssq - (float)CC * mq * mq, 0.f));
        float np = sqrtf(fmaxf(ssp - (float)CC * mp * mp, 0.f));
        muq[n] = mq;  mup[n] = mp;
        ivq[n] = 1.f / fmaxf(nq, 1e-8f);
        ivp[n] = 1.f / fmaxf(np, 1e-8f);
        wr[n] = fmaxf(w1, 0.f) + 0.001f;
        wc[n] = fmaxf(w2, 0.f) + 0.001f;
    }
    __syncthreads();
    if (tid == 0) {
        float s1 = 0.f, s2 = 0.f;
        for (int n = 0; n < NN; ++n) { s1 += wr[n]; s2 += wc[n]; }
        sums[0] = s1; sums[1] = s2;
    }
    __syncthreads();
    if (tid < NN) {
        logr[(size_t)b * NN + tid] = __logf(wr[tid] / sums[0]);
        logc[(size_t)b * NN + tid] = __logf(wc[tid] / sums[1]);
    }
    // Transposed (node-major) bf16 write: e -> (c = e/NN, n = e%NN), reads coalesced
    for (int e = tid; e < CC * NN; e += blockDim.x) {
        int n = e % NN, c = e / NN;
        float qv = query[(baseBC + c) * NN + n];
        float pv = proto[(baseBC + c) * NN + n];
        qnT[((size_t)b * NN + n) * CC + c] = f2bf((qv - muq[n]) * ivq[n]);
        pnT[((size_t)b * NN + n) * CC + c] = f2bf((pv - mup[n]) * ivp[n]);
    }
}

// ---------------------------------------------------------------------------
// K3: batched GEMM  sim[b] = qn[b] (144x512) * pn[b]^T (512x144), bf16 WMMA.
// 3x3 register-blocked: one wave computes a 48x48 output block = 9 independent
// 16x16 accumulators (72 VGPRs). Per k-step: 3 A-frags + 3 B-frags -> 9 WMMAs
// with no cross-dependencies (D->C accumulate only). __launch_bounds__(32,1)
// lifts the VGPR cap (single wave32 per block, no occupancy target) and
// "#pragma unroll 1" keeps the K loop body at one fragment generation so the
// accumulators never spill to scratch.
// ---------------------------------------------------------------------------
__global__ void __launch_bounds__(32, 1)
k_gemm(const unsigned short* __restrict__ qnT,
       const unsigned short* __restrict__ pnT,
       float* __restrict__ sim) {
    const int blk = blockIdx.x;             // b*9 + superTile
    const int b   = blk / 9;
    const int st  = blk % 9;
    const int bi  = st / 3;                 // super-tile row   (48 rows)
    const int bj  = st % 3;                 // super-tile col   (48 cols)
    const int lane = threadIdx.x;           // wave32
    const int half = lane >> 4;             // 0: lanes 0-15, 1: lanes 16-31
    const int l15  = lane & 15;

    // A fragment row (M) and B fragment column (N) both = lane&15
    const unsigned short* qbase = qnT + ((size_t)b * NN + bi * 48 + l15) * CC;
    const unsigned short* pbase = pnT + ((size_t)b * NN + bj * 48 + l15) * CC;
    const int kaLo = half * 8;              // A: elems 0-7  -> K = kaLo..kaLo+7
    const int kaHi = kaLo + 16;             // A: elems 8-15 -> K = kaLo+16..kaLo+23
    const int kb   = half * 8;              // B: elems 0-7 -> K = kb..kb+7, elems 8-15 -> K = kb+16..kb+23

    v8f acc[3][3] = {};
    #pragma unroll 1
    for (int c0 = 0; c0 < CC; c0 += 32) {
        union { v16bf v; uint4 q[2]; } A[3], Bm[3];
        #pragma unroll
        for (int r = 0; r < 3; ++r) {
            const unsigned short* qr = qbase + (size_t)r * 16 * CC + c0;
            const unsigned short* pr = pbase + (size_t)r * 16 * CC + c0;
            A[r].q[0]  = *(const uint4*)(qr + kaLo);
            A[r].q[1]  = *(const uint4*)(qr + kaHi);
            Bm[r].q[0] = *(const uint4*)(pr + kb);
            Bm[r].q[1] = *(const uint4*)(pr + kb + 16);
        }
        #pragma unroll
        for (int r = 0; r < 3; ++r)
            #pragma unroll
            for (int s = 0; s < 3; ++s)
                acc[r][s] = __builtin_amdgcn_wmma_f32_16x16x32_bf16(
                    false, A[r].v, false, Bm[s].v, (short)0, acc[r][s], false, false);
    }

    // C/D layout: lanes 0-15 -> N=lane, M=v; lanes 16-31 -> N=lane-16, M=v+8
    #pragma unroll
    for (int r = 0; r < 3; ++r) {
        const int i0 = bi * 48 + r * 16 + half * 8;
        #pragma unroll
        for (int s = 0; s < 3; ++s) {
            const int j = bj * 48 + s * 16 + l15;
            float* out = sim + ((size_t)b * NN + i0) * NN + j;
            #pragma unroll
            for (int v = 0; v < 8; ++v) out[(size_t)v * NN] = acc[r][s][v];
        }
    }
}

// ---------------------------------------------------------------------------
// K4: log-domain Sinkhorn fully in LDS (81KB sim tile, pitch 145 = conflict
// free for both row and column sweeps), then logits = sum(sim*flow)*T/N.
// One workgroup (256 threads) per batch.
// ---------------------------------------------------------------------------
#define PITCH 145
__global__ void k_sinkhorn(const float* __restrict__ sim,
                           const float* __restrict__ logr,
                           const float* __restrict__ logc,
                           float* __restrict__ out) {
    extern __shared__ float lds[];
    const int b = blockIdx.x, tid = threadIdx.x;
    float* S  = lds;                    // NN * PITCH
    float* fv = S + NN * PITCH;         // NN
    float* gv = fv + NN;                // NN
    float* lr = gv + NN;                // NN
    float* lc = lr + NN;                // NN
    float* red = lc + NN;               // blockDim.x

    const float* simb = sim + (size_t)b * NN * NN;
    for (int e = tid; e < NN * NN; e += blockDim.x)
        S[(e / NN) * PITCH + (e % NN)] = simb[e];
    if (tid < NN) {
        fv[tid] = 0.f; gv[tid] = 0.f;
        lr[tid] = logr[(size_t)b * NN + tid];
        lc[tid] = logc[(size_t)b * NN + tid];
    }
    __syncthreads();

    const float invE = 1.f / SINK_EPS;
    for (int it = 0; it < SINK_ITERS; ++it) {
        if (tid < NN) {                             // f update: rows, lse over j
            const int i = tid;
            const float fi = fv[i];
            float m = -3.4e38f, s = 0.f;
            for (int j = 0; j < NN; ++j) {
                float x = (fi + gv[j] - (1.f - S[i * PITCH + j])) * invE;
                if (x > m) { s = s * __expf(m - x) + 1.f; m = x; }
                else       { s += __expf(x - m); }
            }
            fv[i] = fi + SINK_EPS * (lr[i] - (m + __logf(s)));
        }
        __syncthreads();
        if (tid < NN) {                             // g update: cols, lse over i
            const int j = tid;
            const float gj = gv[j];
            float m = -3.4e38f, s = 0.f;
            for (int i = 0; i < NN; ++i) {
                float x = (fv[i] + gj - (1.f - S[i * PITCH + j])) * invE;
                if (x > m) { s = s * __expf(m - x) + 1.f; m = x; }
                else       { s += __expf(x - m); }
            }
            gv[j] = gj + SINK_EPS * (lc[j] - (m + __logf(s)));
        }
        __syncthreads();
    }

    float local = 0.f;
    for (int e = tid; e < NN * NN; e += blockDim.x) {
        int i = e / NN, j = e % NN;
        float s = S[i * PITCH + j];
        local += s * __expf((fv[i] + gv[j] - (1.f - s)) * invE);
    }
    red[tid] = local;
    __syncthreads();
    for (int st = blockDim.x >> 1; st > 0; st >>= 1) {
        if (tid < st) red[tid] += red[tid + st];
        __syncthreads();
    }
    if (tid == 0) out[b] = red[0] * (TEMPERATURE / (float)NN);
}

// ---------------------------------------------------------------------------
extern "C" void kernel_launch(void* const* d_in, const int* in_sizes, int n_in,
                              void* d_out, int out_size, void* d_ws, size_t ws_size,
                              hipStream_t stream) {
    (void)in_sizes; (void)n_in; (void)out_size; (void)ws_size;
    const float* proto = (const float*)d_in[0];
    const float* query = (const float*)d_in[1];
    float* out = (float*)d_out;
    char* ws = (char*)d_ws;

    // Workspace layout (bytes, all 4KB-aligned): ~98 MB total
    const size_t SZ_MEAN = (size_t)BB * CC * sizeof(float);              // 512 KB
    const size_t SZ_LOG  = (size_t)BB * NN * sizeof(float);              // 144 KB
    const size_t SZ_FEAT = (size_t)BB * NN * CC * sizeof(unsigned short);// 36 MB
    size_t off = 0;
    float* pmean = (float*)(ws + off);                 off += SZ_MEAN;
    float* qmean = (float*)(ws + off);                 off += SZ_MEAN;
    float* logr  = (float*)(ws + off);                 off += SZ_LOG;
    float* logc  = (float*)(ws + off);                 off += SZ_LOG;
    unsigned short* qnT = (unsigned short*)(ws + off); off += SZ_FEAT;
    unsigned short* pnT = (unsigned short*)(ws + off); off += SZ_FEAT;
    float* sim   = (float*)(ws + off);                 // B*N*N f32 = 20.25 MB

    k_means<<<(BB * CC + 255) / 256, 256, 0, stream>>>(proto, query, pmean, qmean);
    k_prep<<<BB, 256, 0, stream>>>(proto, query, pmean, qmean, qnT, pnT, logr, logc);
    k_gemm<<<BB * 9, 32, 0, stream>>>(qnT, pnT, sim);

    const size_t ldsBytes = (size_t)(NN * PITCH + 4 * NN + 256) * sizeof(float); // ~85 KB
    k_sinkhorn<<<BB, 256, ldsBytes, stream>>>(sim, logr, logc, out);
}
